// ConstellationGNN_11055245820056
// MI455X (gfx1250) — compile-verified, matrix operand order checked by reading, not run on hardware
//
#include <hip/hip_runtime.h>
#include <hip/hip_bf16.h>

#define HCH 128
#define NH  4
#define CD  32
#define NL  4
#define NCL 88
#define NB  64
#define NEG_SLOPE 0.2f
#define LN_EPS 1e-5f

typedef __attribute__((ext_vector_type(16))) _Float16 v16h;
typedef __attribute__((ext_vector_type(8)))  float    v8f;

static __device__ __forceinline__ int f32_ord(float f) {
    int i = __float_as_int(f);
    return (i >= 0) ? i : (i ^ 0x7FFFFFFF);
}
static __device__ __forceinline__ float ord_f32(int i) {
    int j = (i >= 0) ? i : (i ^ 0x7FFFFFFF);
    return __int_as_float(j);
}

// ---------------------------------------------------------------------------
// Node encoder: h = relu(LayerNorm(x @ W_ne + b_ne))   [one block per node]
// ---------------------------------------------------------------------------
__global__ __launch_bounds__(128) void k_node_encoder(
    const float* __restrict__ x, const float* __restrict__ W,
    const float* __restrict__ b, const float* __restrict__ g,
    const float* __restrict__ be, float* __restrict__ h)
{
    int n = blockIdx.x, c = threadIdx.x;
    float x0 = x[n * 4 + 0], x1 = x[n * 4 + 1], x2 = x[n * 4 + 2], x3 = x[n * 4 + 3];
    float v = b[c] + x0 * W[0 * HCH + c] + x1 * W[1 * HCH + c]
                   + x2 * W[2 * HCH + c] + x3 * W[3 * HCH + c];
    __shared__ float red[128];
    red[c] = v; __syncthreads();
    for (int s = 64; s > 0; s >>= 1) { if (c < s) red[c] += red[c + s]; __syncthreads(); }
    float mu = red[0] * (1.0f / HCH); __syncthreads();
    float d = v - mu;
    red[c] = d * d; __syncthreads();
    for (int s = 64; s > 0; s >>= 1) { if (c < s) red[c] += red[c + s]; __syncthreads(); }
    float var = red[0] * (1.0f / HCH);
    float y = d * rsqrtf(var + LN_EPS) * g[c] + be[c];
    h[(size_t)n * HCH + c] = fmaxf(y, 0.0f);
}

// ---------------------------------------------------------------------------
// Layer init: h_agg = 0, amax = -inf(ordered-int), den = 0
// ---------------------------------------------------------------------------
__global__ void k_init_layer(float* __restrict__ agg, int n_agg,
                             int* __restrict__ amax, float* __restrict__ den, int n4)
{
    int t = blockIdx.x * blockDim.x + threadIdx.x;
    if (t < n_agg) agg[t] = 0.0f;
    if (t < n4) { amax[t] = (int)0x80000000; den[t] = 0.0f; }
}

__global__ void k_zero(float* __restrict__ p, int n)
{
    int t = blockIdx.x * blockDim.x + threadIdx.x;
    if (t < n) p[t] = 0.0f;
}

// ---------------------------------------------------------------------------
// WMMA GEMM: Y[M,128] = X[M,128] @ W[128,128]  (f16 inputs, f32 accumulate)
// 256 threads = 8 waves; each wave owns 16 rows x full 128 cols.
// W staged in LDS as f16 B-fragments (16x16x32 layout), 32 KB.
// ---------------------------------------------------------------------------
__global__ __launch_bounds__(256) void k_wmma_gemm128(
    const float* __restrict__ X, const float* __restrict__ W,
    float* __restrict__ Y, int M)
{
    __shared__ __attribute__((aligned(32))) _Float16 Wlds[8][4][32][16];

    int tid = threadIdx.x;
    // Cooperative weight stage: 8 n-tiles x 4 k-steps x 32 lanes, 16 halfs each.
    for (int sidx = tid; sidx < 1024; sidx += 256) {
        int t    = sidx >> 7;         // n-tile 0..7
        int s    = (sidx >> 5) & 3;   // k-step 0..3
        int lane = sidx & 31;
        int col  = lane & 15;
        int kb   = ((lane >> 4) << 3) + (s << 5);
        _Float16* dp = &Wlds[t][s][lane][0];
#pragma unroll
        for (int i = 0; i < 16; ++i) {
            int koff = (i < 8) ? i : (i + 8);  // two contiguous 8-runs: kb.., kb+16..
            dp[i] = (_Float16)W[(size_t)(kb + koff) * HCH + (t * 16 + col)];
        }
    }
    __syncthreads();

    int wave = tid >> 5;
    int lane = tid & 31;
    int row_base = blockIdx.x * 128 + wave * 16;

    int row = row_base + (lane & 15);
    if (row >= M) row = M - 1;               // clamp: keep EXEC all-ones for WMMA
    int kb_lane = (lane >> 4) << 3;

    // A fragments: lane holds row, K runs [kb..kb+7] and [kb+16..kb+23] per k-step.
    v16h afr[4];
#pragma unroll
    for (int s = 0; s < 4; ++s) {
        const float* xp = X + (size_t)row * HCH + (s << 5) + kb_lane;
        v16h a;
#pragma unroll
        for (int i = 0; i < 8; ++i)  a[i] = (_Float16)xp[i];
#pragma unroll
        for (int i = 0; i < 8; ++i)  a[8 + i] = (_Float16)xp[16 + i];
        afr[s] = a;
    }

    v8f acc[8];
#pragma unroll
    for (int t = 0; t < 8; ++t) { v8f z = {}; acc[t] = z; }

#pragma unroll
    for (int t = 0; t < 8; ++t) {
#pragma unroll
        for (int s = 0; s < 4; ++s) {
            v16h bf = *(const v16h*)&Wlds[t][s][lane][0];
            acc[t] = __builtin_amdgcn_wmma_f32_16x16x32_f16(
                false, afr[s], false, bf, (short)0, acc[t], false, false);
        }
    }

    // Store D: lanes 0-15 -> M=vgpr, N=lane; lanes 16-31 -> M=8+vgpr, N=lane-16.
    int mbase = (lane >> 4) << 3;
    int ncol  = lane & 15;
#pragma unroll
    for (int t = 0; t < 8; ++t) {
#pragma unroll
        for (int v = 0; v < 8; ++v) {
            int r = row_base + mbase + v;
            if (r < M) Y[(size_t)r * HCH + t * 16 + ncol] = acc[t][v];
        }
    }
}

// ---------------------------------------------------------------------------
// Per-(node,head) attention coefficients from projected features.
// ---------------------------------------------------------------------------
__global__ void k_head_dots(const float* __restrict__ Hp,
                            const float* __restrict__ avs, const float* __restrict__ avd,
                            float* __restrict__ a_src, float* __restrict__ a_dst, int N)
{
    int t = blockIdx.x * blockDim.x + threadIdx.x;
    if (t >= N * NH) return;
    int n = t >> 2, hh = t & 3;
    const float* hp = Hp + (size_t)n * HCH + hh * CD;
    const float* vs = avs + hh * CD;
    const float* vd = avd + hh * CD;
    float s1 = 0.f, s2 = 0.f;
#pragma unroll
    for (int c = 0; c < CD; ++c) { float hv = hp[c]; s1 = fmaf(hv, vs[c], s1); s2 = fmaf(hv, vd[c], s2); }
    a_src[t] = s1; a_dst[t] = s2;
}

// ---------------------------------------------------------------------------
// Fold edge projection into the attention functional: Pe[k,h] = sum_c We[k,h*32+c]*att_e[h,c]
// ---------------------------------------------------------------------------
__global__ __launch_bounds__(512) void k_compute_Pe(const float* __restrict__ We,
                                                    const float* __restrict__ ae,
                                                    float* __restrict__ Pe)
{
    int t = threadIdx.x;                 // 512 = 128 * 4
    int k = t >> 2, hh = t & 3;
    float s = 0.f;
#pragma unroll
    for (int c = 0; c < CD; ++c) s = fmaf(We[(size_t)k * HCH + hh * CD + c], ae[hh * CD + c], s);
    Pe[k * NH + hh] = s;
}

// ---------------------------------------------------------------------------
// Per-edge: recompute ef on the fly, alpha = leaky(a_src[s]+a_dst[d]+ef@Pe),
// segment-max into amax via ordered-int atomicMax.
// ---------------------------------------------------------------------------
__global__ void k_edge_alpha(const float* __restrict__ ea,
                             const int* __restrict__ src, const int* __restrict__ dst,
                             const float* __restrict__ Wee, const float* __restrict__ bee,
                             const float* __restrict__ Pe,
                             const float* __restrict__ a_src, const float* __restrict__ a_dst,
                             float* __restrict__ alpha, int* __restrict__ amax, int E)
{
    int e = blockIdx.x * blockDim.x + threadIdx.x;
    if (e >= E) return;
    float a = ea[e];
    float acc[4] = {0.f, 0.f, 0.f, 0.f};
#pragma unroll 4
    for (int c = 0; c < HCH; ++c) {
        float v = fmaf(a, Wee[c], bee[c]);
        v = fmaxf(v, 0.0f);
#pragma unroll
        for (int hh = 0; hh < 4; ++hh) acc[hh] = fmaf(v, Pe[c * NH + hh], acc[hh]);
    }
    int sn = src[e], dn = dst[e];
#pragma unroll
    for (int hh = 0; hh < 4; ++hh) {
        float al = a_src[sn * NH + hh] + a_dst[dn * NH + hh] + acc[hh];
        al = (al >= 0.f) ? al : NEG_SLOPE * al;
        alpha[(size_t)e * NH + hh] = al;
        atomicMax(&amax[dn * NH + hh], f32_ord(al));
    }
}

// ---------------------------------------------------------------------------
// ex = exp(alpha - amax[dst]); den[dst] += ex
// ---------------------------------------------------------------------------
__global__ void k_edge_exp(const int* __restrict__ dst,
                           const float* __restrict__ alpha, const int* __restrict__ amax,
                           float* __restrict__ ex, float* __restrict__ den, int E)
{
    int e = blockIdx.x * blockDim.x + threadIdx.x;
    if (e >= E) return;
    int dn = dst[e];
#pragma unroll
    for (int hh = 0; hh < 4; ++hh) {
        float m = ord_f32(amax[dn * NH + hh]);
        float v = __expf(alpha[(size_t)e * NH + hh] - m);
        ex[(size_t)e * NH + hh] = v;
        atomicAdd(&den[dn * NH + hh], v);
    }
}

// ---------------------------------------------------------------------------
// Message aggregation: agg[dst] += (ex/den[dst]) * Hp[src]  (hardware f32 atomics)
// One thread per (edge, channel-within-head); covers all 4 heads.
// ---------------------------------------------------------------------------
__global__ void k_edge_aggregate(const int* __restrict__ src, const int* __restrict__ dst,
                                 const float* __restrict__ ex, const float* __restrict__ den,
                                 const float* __restrict__ Hp, float* __restrict__ agg, int E)
{
    int t = blockIdx.x * blockDim.x + threadIdx.x;
    if (t >= E * 32) return;
    int e = t >> 5, c0 = t & 31;
    int sn = src[e], dn = dst[e];
#pragma unroll
    for (int hh = 0; hh < 4; ++hh) {
        float w = ex[(size_t)e * NH + hh] / den[dn * NH + hh];
        int c = hh * CD + c0;
        atomicAdd(&agg[(size_t)dn * HCH + c], w * Hp[(size_t)sn * HCH + c]);
    }
}

// ---------------------------------------------------------------------------
// Finalize: h += relu(LayerNorm(agg + bias))   [one block per node]
// ---------------------------------------------------------------------------
__global__ __launch_bounds__(128) void k_finalize(
    const float* __restrict__ agg, const float* __restrict__ bias,
    const float* __restrict__ g, const float* __restrict__ be,
    float* __restrict__ h)
{
    int n = blockIdx.x, c = threadIdx.x;
    float v = agg[(size_t)n * HCH + c] + bias[c];
    __shared__ float red[128];
    red[c] = v; __syncthreads();
    for (int s = 64; s > 0; s >>= 1) { if (c < s) red[c] += red[c + s]; __syncthreads(); }
    float mu = red[0] * (1.0f / HCH); __syncthreads();
    float d = v - mu;
    red[c] = d * d; __syncthreads();
    for (int s = 64; s > 0; s >>= 1) { if (c < s) red[c] += red[c + s]; __syncthreads(); }
    float var = red[0] * (1.0f / HCH);
    float y = d * rsqrtf(var + LN_EPS) * g[c] + be[c];
    h[(size_t)n * HCH + c] += fmaxf(y, 0.0f);
}

// ---------------------------------------------------------------------------
// Global mean pool (atomic accumulate per graph)
// ---------------------------------------------------------------------------
__global__ void k_pool(const float* __restrict__ h, const int* __restrict__ batch,
                       float* __restrict__ sums, float* __restrict__ cnts, int N)
{
    int t = blockIdx.x * blockDim.x + threadIdx.x;
    if (t >= N * HCH) return;
    int n = t >> 7, c = t & 127;
    int b = batch[n];
    atomicAdd(&sums[b * HCH + c], h[t]);
    if (c == 0) atomicAdd(&cnts[b], 1.0f);
}

// ---------------------------------------------------------------------------
// Classifier: out = relu(pooled @ W1 + b1) @ W2 + b2   [one block per graph]
// ---------------------------------------------------------------------------
__global__ __launch_bounds__(128) void k_classifier(
    const float* __restrict__ sums, const float* __restrict__ cnts,
    const float* __restrict__ W1, const float* __restrict__ b1,
    const float* __restrict__ W2, const float* __restrict__ b2,
    float* __restrict__ out)
{
    int b = blockIdx.x, t = threadIdx.x;
    __shared__ float pooled[HCH];
    __shared__ float z[64];
    float cnt = fmaxf(cnts[b], 1.0f);
    pooled[t] = sums[b * HCH + t] / cnt;
    __syncthreads();
    if (t < 64) {
        float s = b1[t];
#pragma unroll 4
        for (int c = 0; c < HCH; ++c) s = fmaf(pooled[c], W1[c * 64 + t], s);
        z[t] = fmaxf(s, 0.0f);
    }
    __syncthreads();
    if (t < NCL) {
        float s = b2[t];
#pragma unroll 4
        for (int j = 0; j < 64; ++j) s = fmaf(z[j], W2[j * NCL + t], s);
        out[b * NCL + t] = s;
    }
}

// ---------------------------------------------------------------------------
static inline int grid1(long n, int blk) { return (int)((n + blk - 1) / blk); }

extern "C" void kernel_launch(void* const* d_in, const int* in_sizes, int n_in,
                              void* d_out, int out_size, void* d_ws, size_t ws_size,
                              hipStream_t stream)
{
    const float* x         = (const float*)d_in[0];
    const float* edge_attr = (const float*)d_in[1];
    const int*   ei        = (const int*)d_in[2];
    const int*   batch     = (const int*)d_in[3];
    const float* W_ne  = (const float*)d_in[4];
    const float* b_ne  = (const float*)d_in[5];
    const float* g_ne  = (const float*)d_in[6];
    const float* be_ne = (const float*)d_in[7];
    const float* W_ee  = (const float*)d_in[8];
    const float* b_ee  = (const float*)d_in[9];
    const float* Wl    = (const float*)d_in[10];
    const float* att_s = (const float*)d_in[11];
    const float* att_d = (const float*)d_in[12];
    const float* We    = (const float*)d_in[13];
    const float* att_e = (const float*)d_in[14];
    const float* b_l   = (const float*)d_in[15];
    const float* g_l   = (const float*)d_in[16];
    const float* be_l  = (const float*)d_in[17];
    const float* W1    = (const float*)d_in[18];
    const float* b1    = (const float*)d_in[19];
    const float* W2    = (const float*)d_in[20];
    const float* b2    = (const float*)d_in[21];

    const int N = in_sizes[0] / 4;
    const int E = in_sizes[1];
    const int* src = ei;
    const int* dst = ei + E;

    // Workspace carve-out (floats)
    float* ws = (float*)d_ws;
    float* h      = ws;  ws += (size_t)N * HCH;
    float* Hp     = ws;  ws += (size_t)N * HCH;
    float* agg    = ws;  ws += (size_t)N * HCH;
    float* a_srcB = ws;  ws += (size_t)N * NH;
    float* a_dstB = ws;  ws += (size_t)N * NH;
    int*   amax   = (int*)ws; ws += (size_t)N * NH;
    float* den    = ws;  ws += (size_t)N * NH;
    float* alpha  = ws;  ws += (size_t)E * NH;
    float* ex     = ws;  ws += (size_t)E * NH;
    float* Pe     = ws;  ws += HCH * NH;
    float* sums   = ws;  ws += NB * HCH;   // sums + cnts contiguous for one zero pass
    float* cnts   = ws;  ws += NB;

    k_node_encoder<<<N, 128, 0, stream>>>(x, W_ne, b_ne, g_ne, be_ne, h);

    for (int l = 0; l < NL; ++l) {
        const float* Wl_l  = Wl    + (size_t)l * HCH * HCH;
        const float* We_l  = We    + (size_t)l * HCH * HCH;
        const float* as_l  = att_s + (size_t)l * NH * CD;
        const float* ad_l  = att_d + (size_t)l * NH * CD;
        const float* ae_l  = att_e + (size_t)l * NH * CD;
        const float* bl_l  = b_l   + (size_t)l * HCH;
        const float* gl_l  = g_l   + (size_t)l * HCH;
        const float* bel_l = be_l  + (size_t)l * HCH;

        k_init_layer<<<grid1((long)N * HCH, 256), 256, 0, stream>>>(agg, N * HCH, amax, den, N * NH);
        k_wmma_gemm128<<<grid1(N, 128), 256, 0, stream>>>(h, Wl_l, Hp, N);
        k_head_dots<<<grid1((long)N * NH, 256), 256, 0, stream>>>(Hp, as_l, ad_l, a_srcB, a_dstB, N);
        k_compute_Pe<<<1, 512, 0, stream>>>(We_l, ae_l, Pe);
        k_edge_alpha<<<grid1(E, 256), 256, 0, stream>>>(edge_attr, src, dst, W_ee, b_ee, Pe,
                                                        a_srcB, a_dstB, alpha, amax, E);
        k_edge_exp<<<grid1(E, 256), 256, 0, stream>>>(dst, alpha, amax, ex, den, E);
        k_edge_aggregate<<<grid1((long)E * 32, 256), 256, 0, stream>>>(src, dst, ex, den, Hp, agg, E);
        k_finalize<<<N, 128, 0, stream>>>(agg, bl_l, gl_l, bel_l, h);
    }

    k_zero<<<grid1(NB * HCH + NB, 256), 256, 0, stream>>>(sums, NB * HCH + NB);
    k_pool<<<grid1((long)N * HCH, 256), 256, 0, stream>>>(h, batch, sums, cnts, N);
    k_classifier<<<NB, 128, 0, stream>>>(sums, cnts, W1, b1, W2, b2, (float*)d_out);
}
